// PAINN_30571577213715
// MI455X (gfx1250) — matrix operand order repaired and checked by reading, not compiled, not run on hardware
//
#include <hip/hip_runtime.h>
#include <hip/hip_bf16.h>
#include <stdint.h>

// ---------------------------------------------------------------------------
// PaiNN forward (ROUNDS=1) for MI455X / gfx1250, wave32 WMMA.
// Edge MLP ~21 GFMA dominates -> V_WMMA_F32_16X16X32_F16 with f32 accum.
// Dead work removed: m1 (x zero state_vec) and a_vv (never reaches output),
// so layer-2 GEMMs only compute N-tiles 8..23. Filter W fused (never 492MB).
// Sender-state gather staged via GLOBAL_LOAD_ASYNC_TO_LDS_B128 (ASYNCcnt).
// Layer-2 results scatter straight from WMMA accumulators via f32 atomics.
// ---------------------------------------------------------------------------

#define STATE_DIM 128
#define N_RBF     20
#define N_NODES   10000
#define N_EDGES   320000
#define N_GRAPHS  64
#define PI_F      3.14159265358979f
#define RCUT_INV  (PI_F / 10.0f)

typedef __attribute__((ext_vector_type(16))) _Float16 v16h;
typedef __attribute__((ext_vector_type(8)))  float    v8f;

union Frag16 { v16h v; uint4 q[2]; };

// A-matrix (16x32 f16, MxK) fragment from LDS row-major tile.
// Lane l: row = l&15, h = l>>4. Elements 0-15: K = 16h..16h+15
// (contiguous order, matching packed B).
__device__ inline v16h load_a_frag(const _Float16* sA, int row_stride_h,
                                   int kt, int lane) {
  const int row = lane & 15;
  const int h   = lane >> 4;
  const _Float16* base = sA + row * row_stride_h + kt * 32 + h * 16;
  Frag16 f;
  f.q[0] = *(const uint4*)(base);
  f.q[1] = *(const uint4*)(base + 8);
  return f.v;
}

// B-matrix (32x16 f16, KxN) fragment from pre-packed weights:
// packed idx = (((kt*NT)+nt)*32 + lane)*16 + i ; k = (lane>>4)*16+i,
// n = lane&15. 32B contiguous per lane -> coalesced b128 pairs from L2.
__device__ inline v16h load_b_frag(const _Float16* wp, int NT, int kt, int nt,
                                   int lane) {
  const _Float16* base = wp + ((size_t)((kt * NT + nt) * 32 + lane)) * 16;
  Frag16 f;
  f.q[0] = *(const uint4*)(base);
  f.q[1] = *(const uint4*)(base + 8);
  return f.v;
}

__device__ inline v8f wmma_f16(v16h a, v16h b, v8f c) {
  return __builtin_amdgcn_wmma_f32_16x16x32_f16(false, a, false, b,
                                                (short)0, c, false, false);
}

__device__ inline float silu(float x) { return x / (1.0f + __expf(-x)); }

// Async global->LDS 16B copy (gfx1250 GLOBAL_LOAD_ASYNC_TO_LDS_B128).
// lds_off: LDS byte offset (low 32 bits of generic LDS address);
// gaddr:   64-bit global address (VGPR pair). Tracked by ASYNCcnt.
__device__ inline void async_copy_b128(unsigned lds_off, const void* gaddr) {
  asm volatile("global_load_async_to_lds_b128 %0, %1, off"
               :: "v"(lds_off), "v"(gaddr) : "memory");
}
__device__ inline void wait_asynccnt0() {
  asm volatile("s_wait_asynccnt 0x0" ::: "memory");
}

// ---------------------------------------------------------------------------
// Pack f32 row-major [K][N] weight into WMMA-B fragment order, f16.
// ---------------------------------------------------------------------------
__global__ void pack_b_kernel(const float* __restrict__ src,
                              _Float16* __restrict__ dst, int K, int N) {
  int tid = blockIdx.x * blockDim.x + threadIdx.x;
  if (tid >= K * N) return;
  int NT   = N >> 4;
  int i    = tid & 15;
  int lane = (tid >> 4) & 31;
  int tile = tid >> 9;            // kt*NT + nt
  int nt   = tile % NT;
  int kt   = tile / NT;
  int k = kt * 32 + ((lane >> 4) << 4) + i;
  int n = nt * 16 + (lane & 15);
  dst[tid] = (_Float16)src[(size_t)k * N + n];
}

// ---------------------------------------------------------------------------
// Init: state1 = embedding[atoms] (f32), s0h = same (f16), svec1=0, gs=0.
// ---------------------------------------------------------------------------
__global__ void init_kernel(const int* __restrict__ atoms,
                            const float* __restrict__ embedding,
                            float* __restrict__ state1,
                            _Float16* __restrict__ s0h,
                            float* __restrict__ svec1,
                            float* __restrict__ gs) {
  size_t idx = (size_t)blockIdx.x * blockDim.x + threadIdx.x;
  if (idx < (size_t)N_NODES * STATE_DIM) {
    int node = (int)(idx >> 7);
    float v = embedding[(size_t)atoms[node] * STATE_DIM + (idx & 127)];
    state1[idx] = v;
    s0h[idx]    = (_Float16)v;
  }
  if (idx < (size_t)N_NODES * 3 * STATE_DIM) svec1[idx] = 0.0f;
  if (idx < (size_t)N_GRAPHS * STATE_DIM)    gs[idx]    = 0.0f;
}

// ---------------------------------------------------------------------------
// Edge kernel: 64 edges per block (5000 blocks, 8 waves).
//   phi = silu(x@w1+b1)@w2+b2 on WMMA; layer2 only live N-tiles 8..23,
//   both of a wave's N-tiles computed in one pass so each A-fragment
//   ds_load feeds two WMMAs (8 accumulator chains).
//   Epilogue straight from accumulators: pf = phi*W, scatter m2 -> state1,
//   norm (x) m3 -> svec1 via global_atomic_add_f32.
// ---------------------------------------------------------------------------
__global__ void __launch_bounds__(256) edge_kernel(
    const int* __restrict__ senders, const int* __restrict__ receivers,
    const float* __restrict__ edge_vectors, const _Float16* __restrict__ s0h,
    const _Float16* __restrict__ w1p, const _Float16* __restrict__ w2p,
    const float* __restrict__ phi_b1, const float* __restrict__ phi_b2,
    const float* __restrict__ filt_w, const float* __restrict__ filt_b,
    float* __restrict__ state1, float* __restrict__ svec1) {
  __shared__ __align__(16) _Float16 sA[64][128];   // sender states (16KB)
  __shared__ __align__(16) _Float16 sH[64][128];   // silu(layer1)  (16KB)
  __shared__ float sNorm[64][3];
  __shared__ float sRbf[64][N_RBF];
  __shared__ float sCut[64];
  __shared__ int   sSend[64], sRecv[64];

  const int tid  = threadIdx.x;
  const int lane = tid & 31;
  const int wave = tid >> 5;
  const int e0   = blockIdx.x * 64;

  __builtin_prefetch(w1p, 0, 3);                  // weights live in L2
  __builtin_prefetch(w2p, 0, 3);

  if (tid < 64) {
    sSend[tid] = senders[e0 + tid];
    sRecv[tid] = receivers[e0 + tid];
  }
  __syncthreads();

  // gather 64 sender rows (64x128 f16): async global->LDS, 16B/lane x4
#pragma unroll
  for (int it = 0; it < 4; ++it) {
    int idx = tid + it * 256;                     // 1024 x b128
    int e = idx >> 4, seg = idx & 15;
    const uint4* src = (const uint4*)(s0h + (size_t)sSend[e] * STATE_DIM) + seg;
    unsigned lds_off = (unsigned)(uintptr_t)(&sA[e][seg * 8]);
    async_copy_b128(lds_off, (const void*)src);
  }
  if (tid < 64) {
    const float* ev = edge_vectors + (size_t)(e0 + tid) * 3;
    float x = ev[0], y = ev[1], z = ev[2];
    float r = sqrtf(x * x + y * y + z * z);
    float inv = 1.0f / r;
    sNorm[tid][0] = x * inv; sNorm[tid][1] = y * inv; sNorm[tid][2] = z * inv;
#pragma unroll
    for (int i = 0; i < N_RBF; ++i)
      sRbf[tid][i] = __sinf((float)(i + 1) * RCUT_INV * r) * inv;
    sCut[tid] = (r < 10.0f) ? 0.5f * (__cosf(RCUT_INV * r) + 1.0f) : 0.0f;
  }
  wait_asynccnt0();                               // our async copies landed
  __syncthreads();

  const int n    = lane & 15;
  const int mrow = (lane >> 4) * 8;

  // ---- layer 1: [64x128]@[128x128]; wave -> nt, 4 M-tiles per B frag ----
  {
    v8f acc[4];
#pragma unroll
    for (int mt = 0; mt < 4; ++mt) acc[mt] = (v8f){0, 0, 0, 0, 0, 0, 0, 0};
#pragma unroll
    for (int kt = 0; kt < 4; ++kt) {
      v16h b = load_b_frag(w1p, 8, kt, wave, lane);
#pragma unroll
      for (int mt = 0; mt < 4; ++mt) {
        v16h a = load_a_frag(&sA[mt * 16][0], 128, kt, lane);
        acc[mt] = wmma_f16(a, b, acc[mt]);
      }
    }
    float bias = phi_b1[wave * 16 + n];
#pragma unroll
    for (int mt = 0; mt < 4; ++mt)
#pragma unroll
      for (int v = 0; v < 8; ++v)
        sH[mt * 16 + mrow + v][wave * 16 + n] = (_Float16)silu(acc[mt][v] + bias);
  }
  __syncthreads();

  // ---- layer 2: live N-tiles {8+2w, 9+2w}; A frag shared by both ----
  {
    const int nt0 = 8 + wave * 2;
    v8f acc[2][4];
#pragma unroll
    for (int t = 0; t < 2; ++t)
#pragma unroll
      for (int mt = 0; mt < 4; ++mt) acc[t][mt] = (v8f){0, 0, 0, 0, 0, 0, 0, 0};
#pragma unroll
    for (int kt = 0; kt < 4; ++kt) {
      v16h b0 = load_b_frag(w2p, 24, kt, nt0,     lane);
      v16h b1 = load_b_frag(w2p, 24, kt, nt0 + 1, lane);
#pragma unroll
      for (int mt = 0; mt < 4; ++mt) {
        v16h a = load_a_frag(&sH[mt * 16][0], 128, kt, lane);
        acc[0][mt] = wmma_f16(a, b0, acc[0][mt]);
        acc[1][mt] = wmma_f16(a, b1, acc[1][mt]);
      }
    }
#pragma unroll
    for (int t = 0; t < 2; ++t) {
      const int nt = nt0 + t;
      const int j  = nt * 16 + n;                 // 128..383
      const int jj = j & 127;
      const bool to_state = (nt < 16);            // m2 vs m3
      float fb   = filt_b[j];
      float bias = phi_b2[j];
      float wcol[N_RBF];
#pragma unroll
      for (int k = 0; k < N_RBF; ++k) wcol[k] = filt_w[k * 384 + j];
#pragma unroll
      for (int mt = 0; mt < 4; ++mt) {
#pragma unroll
        for (int v = 0; v < 8; ++v) {
          int e = mt * 16 + mrow + v;
          float wsum = fb;
#pragma unroll
          for (int k = 0; k < N_RBF; ++k) wsum += wcol[k] * sRbf[e][k];
          float pf = (acc[t][mt][v] + bias) * wsum * sCut[e];
          int recv = sRecv[e];
          if (to_state) {
            atomicAdd(&state1[(size_t)recv * STATE_DIM + jj], pf);
          } else {
            float* base = svec1 + (size_t)recv * 384 + jj;
            atomicAdd(base + 0,   sNorm[e][0] * pf);
            atomicAdd(base + 128, sNorm[e][1] * pf);
            atomicAdd(base + 256, sNorm[e][2] * pf);
          }
        }
      }
    }
  }
}

// ---------------------------------------------------------------------------
// U/V kernel: 16 nodes/block. Rows laid out [p][node] so each wave holds
// U and V tiles for all 3 p in registers (6 acc chains, 24 WMMAs/wave);
// UV = sum_p U*V and ||V|| reduce lane-locally, no LDS result staging.
// ---------------------------------------------------------------------------
__global__ void __launch_bounds__(256) uv_kernel(
    const float* __restrict__ svec1, const _Float16* __restrict__ Up,
    const _Float16* __restrict__ Vp, float* __restrict__ UVg,
    float* __restrict__ Vng) {
  __shared__ __align__(16) _Float16 sX[48][128];  // row = p*16 + node_local
  const int tid = threadIdx.x, lane = tid & 31, wave = tid >> 5;
  const int n0 = blockIdx.x * 16;

  for (int it = 0; it < 24; ++it) {
    int idx = tid + it * 256;                     // 48*128
    int r = idx >> 7, j = idx & 127;
    int p = r >> 4, nl = r & 15;
    sX[r][j] = (_Float16)svec1[((size_t)(n0 + nl) * 3 + p) * STATE_DIM + j];
  }
  __syncthreads();

  v8f aU[3], aV[3];
#pragma unroll
  for (int p = 0; p < 3; ++p) {
    aU[p] = (v8f){0, 0, 0, 0, 0, 0, 0, 0};
    aV[p] = (v8f){0, 0, 0, 0, 0, 0, 0, 0};
  }
#pragma unroll
  for (int kt = 0; kt < 4; ++kt) {
    v16h bu = load_b_frag(Up, 8, kt, wave, lane);
    v16h bv = load_b_frag(Vp, 8, kt, wave, lane);
#pragma unroll
    for (int p = 0; p < 3; ++p) {
      v16h a = load_a_frag(&sX[p * 16][0], 128, kt, lane);
      aU[p] = wmma_f16(a, bu, aU[p]);
      aV[p] = wmma_f16(a, bv, aV[p]);
    }
  }
  const int n = lane & 15, mrow = (lane >> 4) * 8;
  const int j = wave * 16 + n;
#pragma unroll
  for (int v = 0; v < 8; ++v) {
    size_t node = (size_t)(n0 + mrow + v);
    float u0 = aU[0][v], u1 = aU[1][v], u2 = aU[2][v];
    float v0 = aV[0][v], v1 = aV[1][v], v2 = aV[2][v];
    UVg[node * STATE_DIM + j] = u0 * v0 + u1 * v1 + u2 * v2;
    Vng[node * STATE_DIM + j] = sqrtf(v0 * v0 + v1 * v1 + v2 * v2);
  }
}

// ---------------------------------------------------------------------------
// Update MLP: a = silu(cat(Vnorm,state)@uw1+b1)@uw2+b2.
// a_vv dead -> layer2 N-tiles 8..23 only (A frag shared by the wave's two
// tiles); epilogue from accumulators:
// state1 += a_ss   and   state1 += UV * a_sv  (atomic f32 adds).
// ---------------------------------------------------------------------------
__global__ void __launch_bounds__(256) upd_kernel(
    const float* __restrict__ Vng, const float* __restrict__ UVg,
    const _Float16* __restrict__ uw1p, const _Float16* __restrict__ uw2p,
    const float* __restrict__ upd_b1, const float* __restrict__ upd_b2,
    float* __restrict__ state1) {
  __shared__ __align__(16) _Float16 sA[16][256];  // cat(Vnorm,state) (8KB)
  __shared__ __align__(16) _Float16 sH[16][128];  // silu(layer1)     (4KB)
  const int tid = threadIdx.x, lane = tid & 31, wave = tid >> 5;
  const int n0 = blockIdx.x * 16;
  const int n = lane & 15, mrow = (lane >> 4) * 8;

  __builtin_prefetch(uw1p, 0, 3);
  __builtin_prefetch(uw2p, 0, 3);

  for (int it = 0; it < 16; ++it) {
    int idx = tid + it * 256;                     // 16*256
    int e = idx >> 8, k = idx & 255;
    size_t node = (size_t)(n0 + e);
    float v = (k < 128) ? Vng[node * STATE_DIM + k]
                        : state1[node * STATE_DIM + (k - 128)];
    sA[e][k] = (_Float16)v;
  }
  __syncthreads();

  {                                               // layer 1: K=256, N=128
    v8f acc = {0, 0, 0, 0, 0, 0, 0, 0};
#pragma unroll
    for (int kt = 0; kt < 8; ++kt) {
      v16h a = load_a_frag(&sA[0][0], 256, kt, lane);
      v16h b = load_b_frag(uw1p, 8, kt, wave, lane);
      acc = wmma_f16(a, b, acc);
    }
    float bias = upd_b1[wave * 16 + n];
#pragma unroll
    for (int v = 0; v < 8; ++v)
      sH[mrow + v][wave * 16 + n] = (_Float16)silu(acc[v] + bias);
  }
  __syncthreads();

  {                                               // layer 2: live tiles only
    const int nt0 = 8 + wave * 2;
    v8f acc[2];
    acc[0] = (v8f){0, 0, 0, 0, 0, 0, 0, 0};
    acc[1] = (v8f){0, 0, 0, 0, 0, 0, 0, 0};
#pragma unroll
    for (int kt = 0; kt < 4; ++kt) {
      v16h a  = load_a_frag(&sH[0][0], 128, kt, lane);
      v16h b0 = load_b_frag(uw2p, 24, kt, nt0,     lane);
      v16h b1 = load_b_frag(uw2p, 24, kt, nt0 + 1, lane);
      acc[0] = wmma_f16(a, b0, acc[0]);
      acc[1] = wmma_f16(a, b1, acc[1]);
    }
#pragma unroll
    for (int t = 0; t < 2; ++t) {
      const int nt = nt0 + t;
      const int jj = (nt * 16 + n) & 127;
      const bool is_sv = (nt < 16);
      float bias = upd_b2[nt * 16 + n];
#pragma unroll
      for (int v = 0; v < 8; ++v) {
        size_t node = (size_t)(n0 + mrow + v);
        float aval = acc[t][v] + bias;
        float contrib = is_sv ? UVg[node * STATE_DIM + jj] * aval : aval;
        atomicAdd(&state1[node * STATE_DIM + jj], contrib);
      }
    }
  }
}

// ---------------------------------------------------------------------------
// Segment sum + readout
// ---------------------------------------------------------------------------
__global__ void segsum_kernel(const float* __restrict__ state1,
                              const int* __restrict__ ngi,
                              float* __restrict__ gs) {
  size_t idx = (size_t)blockIdx.x * blockDim.x + threadIdx.x;
  if (idx >= (size_t)N_NODES * STATE_DIM) return;
  int node = (int)(idx >> 7);
  atomicAdd(&gs[(size_t)ngi[node] * STATE_DIM + (idx & 127)], state1[idx]);
}

__global__ void __launch_bounds__(256) readout_kernel(
    const float* __restrict__ gs, const float* __restrict__ ow1,
    const float* __restrict__ ob1, const float* __restrict__ ow2,
    const float* __restrict__ ob2, float* __restrict__ out) {
  __shared__ float sHh[N_GRAPHS][STATE_DIM];
  const int tid = threadIdx.x;
  for (int it = 0; it < 32; ++it) {
    int idx = tid + it * 256;                     // 64*128
    int g = idx >> 7, j = idx & 127;
    float acc = ob1[j];
    for (int k = 0; k < STATE_DIM; ++k)
      acc += gs[g * STATE_DIM + k] * ow1[k * STATE_DIM + j];
    sHh[g][j] = silu(acc);
  }
  __syncthreads();
  if (tid < N_GRAPHS) {
    float o = ob2[0];
    for (int j = 0; j < STATE_DIM; ++j) o += sHh[tid][j] * ow2[j];
    out[tid] = o;
  }
}

// ---------------------------------------------------------------------------
// Launch
// ---------------------------------------------------------------------------
extern "C" void kernel_launch(void* const* d_in, const int* in_sizes, int n_in,
                              void* d_out, int out_size, void* d_ws,
                              size_t ws_size, hipStream_t stream) {
  const int*   node_atoms = (const int*)d_in[0];
  const int*   node_graph = (const int*)d_in[1];
  const int*   senders    = (const int*)d_in[2];
  const int*   receivers  = (const int*)d_in[3];
  const float* edge_vec   = (const float*)d_in[4];
  const float* embedding  = (const float*)d_in[5];
  const float* phi_w1 = (const float*)d_in[6];
  const float* phi_b1 = (const float*)d_in[7];
  const float* phi_w2 = (const float*)d_in[8];
  const float* phi_b2 = (const float*)d_in[9];
  const float* filt_w = (const float*)d_in[10];
  const float* filt_b = (const float*)d_in[11];
  const float* upd_w1 = (const float*)d_in[12];
  const float* upd_b1 = (const float*)d_in[13];
  const float* upd_w2 = (const float*)d_in[14];
  const float* upd_b2 = (const float*)d_in[15];
  const float* U_w    = (const float*)d_in[16];
  const float* V_w    = (const float*)d_in[17];
  const float* out_w1 = (const float*)d_in[18];
  const float* out_b1 = (const float*)d_in[19];
  const float* out_w2 = (const float*)d_in[20];
  const float* out_b2 = (const float*)d_in[21];
  float* out = (float*)d_out;

  // workspace layout (bytes); total ~33.7 MB
  char* ws = (char*)d_ws;
  _Float16* w1p  = (_Float16*)(ws + 0);         //  32768
  _Float16* w2p  = (_Float16*)(ws + 32768);     //  98304
  _Float16* uw1p = (_Float16*)(ws + 131072);    //  65536
  _Float16* uw2p = (_Float16*)(ws + 196608);    //  98304
  _Float16* Up   = (_Float16*)(ws + 294912);    //  32768
  _Float16* Vp   = (_Float16*)(ws + 327680);    //  32768
  _Float16* s0h  = (_Float16*)(ws + 360448);    //  2,560,000
  float* state1  = (float*)(ws + 2920448);      //  5,120,000
  float* svec1   = (float*)(ws + 8040448);      // 15,360,000
  float* UVg     = (float*)(ws + 23400448);     //  5,120,000
  float* Vng     = (float*)(ws + 28520448);     //  5,120,000
  float* gs      = (float*)(ws + 33640448);     //     32,768

  init_kernel<<<15000, 256, 0, stream>>>(node_atoms, embedding, state1, s0h,
                                         svec1, gs);
  pack_b_kernel<<<64,  256, 0, stream>>>(phi_w1, w1p, 128, 128);
  pack_b_kernel<<<192, 256, 0, stream>>>(phi_w2, w2p, 128, 384);
  pack_b_kernel<<<128, 256, 0, stream>>>(upd_w1, uw1p, 256, 128);
  pack_b_kernel<<<192, 256, 0, stream>>>(upd_w2, uw2p, 128, 384);
  pack_b_kernel<<<64,  256, 0, stream>>>(U_w, Up, 128, 128);
  pack_b_kernel<<<64,  256, 0, stream>>>(V_w, Vp, 128, 128);

  edge_kernel<<<N_EDGES / 64, 256, 0, stream>>>(
      senders, receivers, edge_vec, s0h, w1p, w2p, phi_b1, phi_b2, filt_w,
      filt_b, state1, svec1);

  uv_kernel<<<N_NODES / 16, 256, 0, stream>>>(svec1, Up, Vp, UVg, Vng);
  upd_kernel<<<N_NODES / 16, 256, 0, stream>>>(Vng, UVg, uw1p, uw2p, upd_b1,
                                               upd_b2, state1);
  segsum_kernel<<<(N_NODES * STATE_DIM) / 256, 256, 0, stream>>>(state1,
                                                                 node_graph, gs);
  readout_kernel<<<1, 256, 0, stream>>>(gs, out_w1, out_b1, out_w2, out_b2,
                                        out);
}
